// DecompAtt_14731737825592
// MI455X (gfx1250) — matrix operand order repaired and checked by reading, not compiled, not run on hardware
//
#include <hip/hip_runtime.h>

// ---------------------------------------------------------------------------
// DecompAtt forward on gfx1250 (MI455X): all GEMM-shaped ops via
// v_wmma_f32_16x16x32_f16 (wave32), f32 accumulate. B operands staged in LDS
// in fragment order (b128 ds loads), A operands loaded as b128 global loads.
// ---------------------------------------------------------------------------

typedef __attribute__((ext_vector_type(16))) _Float16 v16h;
typedef __attribute__((ext_vector_type(8)))  _Float16 v8h;
typedef __attribute__((ext_vector_type(2)))  _Float16 v2h;
typedef __attribute__((ext_vector_type(8)))  float    v8f;
typedef __attribute__((ext_vector_type(4)))  float    v4f;

#define BB 32
#define PP 384
#define HH 384
#define DD 512
#define SB_STRIDE 40  // halfs per staged column (32 data + 8 pad, 80B, 16B-aligned)

__device__ __forceinline__ v8f wmma16(v16h a, v16h b, v8f c) {
  // D = A(16x32 f16) * B(32x16 f16) + C(16x16 f32)
  return __builtin_amdgcn_wmma_f32_16x16x32_f16(
      /*neg_a=*/false, a, /*neg_b=*/false, b,
      /*c_mod=*/(short)0, c, /*reuse_a=*/false, /*reuse_b=*/false);
}

__device__ __forceinline__ v16h cat8(v8h lo, v8h hi) {
  return __builtin_shufflevector(lo, hi, 0, 1, 2, 3, 4, 5, 6, 7,
                                 8, 9, 10, 11, 12, 13, 14, 15);
}

// A fragment 16x32 from row-major f16 [*, ld]; two b128 global loads.
__device__ __forceinline__ v16h load_a_f16(const _Float16* __restrict__ base, int ld,
                                           int row0, int kb, int lane) {
  const int hs = lane >> 4;
  const _Float16* p = base + (size_t)(row0 + (lane & 15)) * ld + kb + hs * 8;
  const v8h lo = *(const v8h*)p;         // K = kb + hs*8 .. +7
  const v8h hi = *(const v8h*)(p + 16);  // K = kb + hs*8 + 16 .. +7
  return cat8(lo, hi);
}

// A fragment 16x32 from row-major f32; four b128 global loads + cvt.
__device__ __forceinline__ v16h load_a_f32(const float* __restrict__ base, int ld,
                                           int row0, int kb, int lane) {
  const int hs = lane >> 4;
  const float* p = base + (size_t)(row0 + (lane & 15)) * ld + kb + hs * 8;
  const v4f x0 = *(const v4f*)p;
  const v4f x1 = *(const v4f*)(p + 4);
  const v4f x2 = *(const v4f*)(p + 16);
  const v4f x3 = *(const v4f*)(p + 20);
  v16h a;
#pragma unroll
  for (int i = 0; i < 4; ++i) {
    a[i]      = (_Float16)x0[i];
    a[4 + i]  = (_Float16)x1[i];
    a[8 + i]  = (_Float16)x2[i];
    a[12 + i] = (_Float16)x3[i];
  }
  return a;
}

// B fragment 32x16 where B[k][n] = M2[col+n][kb+k] (C = A * M2^T); contiguous
// along K per lane -> two b128 global loads.
__device__ __forceinline__ v16h load_b_rowsT_f16(const _Float16* __restrict__ base, int ld,
                                                 int kb, int col, int lane) {
  const _Float16* p = base + (size_t)(col + (lane & 15)) * ld + kb + (lane >> 4) * 8;
  return cat8(*(const v8h*)p, *(const v8h*)(p + 16));
}

// Stage a 32(k) x 64(n) tile of a k-major f32 matrix into LDS in fragment
// order: sB[n*SB_STRIDE + k] (f16). Coalesced global reads (consecutive
// threads -> consecutive n), packed v2h stores (two k per thread).
__device__ __forceinline__ void stage_b_tile(const float* __restrict__ src, int ld,
                                             int kb, int col0, _Float16* sB, int tid) {
#pragma unroll 2
  for (int i = tid; i < 1024; i += 256) {
    const int n = i & 63;
    const int kk = i >> 6;  // 0..15, covers k = 2*kk, 2*kk+1
    const float lo = src[(size_t)(kb + 2 * kk) * ld + col0 + n];
    const float hi = src[(size_t)(kb + 2 * kk + 1) * ld + col0 + n];
    const v2h pk = {(_Float16)lo, (_Float16)hi};
    *(v2h*)(sB + n * SB_STRIDE + 2 * kk) = pk;
  }
}

// B fragment 32x16 from the staged LDS tile: two b128 ds loads per lane.
__device__ __forceinline__ v16h load_b_staged(const _Float16* sB, int colLocal, int lane) {
  const _Float16* p = sB + (colLocal + (lane & 15)) * SB_STRIDE + (lane >> 4) * 8;
  return cat8(*(const v8h*)p, *(const v8h*)(p + 16));
}

// ---------------------------------------------------------------------------
// K1: proj = relu(X @ W_attend + b)  X:[12288,512] f32 -> Y:[12288,512] f16
// grid (8, 96, 2) x 256 threads; per-wave 16x64 tile; W staged in LDS.
// ---------------------------------------------------------------------------
__global__ void __launch_bounds__(256) attend_gemm_kernel(
    const float* __restrict__ Xp, const float* __restrict__ Xh,
    const float* __restrict__ W, const float* __restrict__ bias,
    _Float16* __restrict__ Yp, _Float16* __restrict__ Yh) {
  const float* X = (blockIdx.z == 0) ? Xp : Xh;
  _Float16* Y = (blockIdx.z == 0) ? Yp : Yh;
  const int lane = threadIdx.x & 31;
  const int wave = threadIdx.x >> 5;
  const int row0 = blockIdx.y * 128 + wave * 16;
  const int col0 = blockIdx.x * 64;
  __shared__ __align__(16) _Float16 sB[64 * SB_STRIDE];
  const v8f zero = {};
  v8f acc[4] = {zero, zero, zero, zero};

  for (int kb = 0; kb < DD; kb += 32) {
    __syncthreads();
    stage_b_tile(W, DD, kb, col0, sB, threadIdx.x);
    __syncthreads();
    if (kb + 32 < DD)
      __builtin_prefetch(X + (size_t)(row0 + (lane & 15)) * DD + kb + 32, 0, 1);
    const v16h a = load_a_f32(X, DD, row0, kb, lane);
#pragma unroll
    for (int t = 0; t < 4; ++t)
      acc[t] = wmma16(a, load_b_staged(sB, t * 16, lane), acc[t]);
  }
  const int hs = lane >> 4, nloc = lane & 15;
#pragma unroll
  for (int t = 0; t < 4; ++t) {
    const int col = col0 + t * 16 + nloc;
    const float bv = bias[col];
#pragma unroll
    for (int r = 0; r < 8; ++r) {
      const int row = row0 + r + hs * 8;
      Y[(size_t)row * DD + col] = (_Float16)fmaxf(acc[t][r] + bv, 0.0f);
    }
  }
}

// ---------------------------------------------------------------------------
// K2: sim[b] = projP[b] @ projH[b]^T   [384,512]x[384,512]^T -> [384,384] f32
// grid (6, 3, 32) x 256 threads; pure b128 global traffic, no LDS needed.
// ---------------------------------------------------------------------------
__global__ void __launch_bounds__(256) sim_gemm_kernel(
    const _Float16* __restrict__ Pp, const _Float16* __restrict__ Ph,
    float* __restrict__ sim) {
  const int batch = blockIdx.z;
  const int lane = threadIdx.x & 31;
  const int wave = threadIdx.x >> 5;
  const int row0 = blockIdx.y * 128 + wave * 16;  // p
  const int col0 = blockIdx.x * 64;               // h
  const _Float16* A = Pp + (size_t)batch * PP * DD;
  const _Float16* Bm = Ph + (size_t)batch * HH * DD;
  float* O = sim + (size_t)batch * PP * HH;
  const v8f zero = {};
  v8f acc[4] = {zero, zero, zero, zero};

  for (int kb = 0; kb < DD; kb += 32) {
    const v16h a = load_a_f16(A, DD, row0, kb, lane);
#pragma unroll
    for (int t = 0; t < 4; ++t)
      acc[t] = wmma16(a, load_b_rowsT_f16(Bm, DD, kb, col0 + t * 16, lane), acc[t]);
  }
  const int hs = lane >> 4, nloc = lane & 15;
#pragma unroll
  for (int t = 0; t < 4; ++t)
#pragma unroll
    for (int r = 0; r < 8; ++r)
      O[(size_t)(row0 + r + hs * 8) * HH + col0 + t * 16 + nloc] = acc[t][r];
}

// ---------------------------------------------------------------------------
// K3: AllenNLP legacy masked softmax, one block per output row (len 384).
// transposed==0: rows of sim (p2h); transposed==1: cols of sim (h2p).
// Writes f32 to final output and an f16 copy for the next GEMM.
// ---------------------------------------------------------------------------
__global__ void __launch_bounds__(128) masked_softmax_kernel(
    const float* __restrict__ sim, const int* __restrict__ mask,
    float* __restrict__ out_f32, _Float16* __restrict__ out_f16, int transposed) {
  const int rid = blockIdx.x;  // b*384 + (p or h)
  const int b = rid / 384;
  const int r = rid - b * 384;
  const int tid = threadIdx.x, lane = tid & 31, wid = tid >> 5;
  const float* srow;
  int stride;
  if (!transposed) { srow = sim + (size_t)rid * HH; stride = 1; }
  else             { srow = sim + (size_t)b * PP * HH + r; stride = HH; }
  const int* mrow = mask + b * 384;

  float t0[3], mk[3];
  float mx = -3.4e38f;
#pragma unroll
  for (int c = 0; c < 3; ++c) {
    const int j = tid + c * 128;
    mk[c] = (float)mrow[j];
    t0[c] = srow[(size_t)j * stride] * mk[c];
    mx = fmaxf(mx, t0[c]);
  }
#pragma unroll
  for (int off = 16; off; off >>= 1) mx = fmaxf(mx, __shfl_xor(mx, off, 32));
  __shared__ float rsm[4], rs1[4], rs2[4];
  if (lane == 0) rsm[wid] = mx;
  __syncthreads();
  mx = fmaxf(fmaxf(rsm[0], rsm[1]), fmaxf(rsm[2], rsm[3]));

  float e[3], s1 = 0.0f, s2 = 0.0f;
#pragma unroll
  for (int c = 0; c < 3; ++c) {
    e[c] = __expf(t0[c] - mx);
    s1 += e[c];
    s2 += e[c] * mk[c];
  }
#pragma unroll
  for (int off = 16; off; off >>= 1) {
    s1 += __shfl_xor(s1, off, 32);
    s2 += __shfl_xor(s2, off, 32);
  }
  if (lane == 0) { rs1[wid] = s1; rs2[wid] = s2; }
  __syncthreads();
  s1 = rs1[0] + rs1[1] + rs1[2] + rs1[3];
  s2 = rs2[0] + rs2[1] + rs2[2] + rs2[3];
  const float inv1 = 1.0f / s1;
  const float dinv = 1.0f / (s2 * inv1 + 1e-13f);
#pragma unroll
  for (int c = 0; c < 3; ++c) {
    const int j = tid + c * 128;
    const float v = e[c] * inv1 * mk[c] * dinv;
    out_f32[(size_t)rid * 384 + j] = v;
    out_f16[(size_t)rid * 384 + j] = (_Float16)v;
  }
}

// ---------------------------------------------------------------------------
// K4: attended = attn @ E   A:[12288,384] f16, E f32 [b][384][512] -> f16.
// grid (8, 96, 2) x 256; E tile (shared by all 8 waves, 128-row blocks never
// straddle a batch) staged through LDS in fragment order.
// ---------------------------------------------------------------------------
__global__ void __launch_bounds__(256) attn_gemm_kernel(
    const _Float16* __restrict__ Ap2h, const _Float16* __restrict__ Ah2p,
    const float* __restrict__ Ep, const float* __restrict__ Eh,
    _Float16* __restrict__ OattH, _Float16* __restrict__ OattP) {
  const _Float16* A;
  const float* E;
  _Float16* O;
  if (blockIdx.z == 0) { A = Ap2h; E = Eh; O = OattH; }
  else                 { A = Ah2p; E = Ep; O = OattP; }
  const int lane = threadIdx.x & 31;
  const int wave = threadIdx.x >> 5;
  const int row0 = blockIdx.y * 128 + wave * 16;
  const int col0 = blockIdx.x * 64;
  const int batch = row0 / 384;
  const float* Eb = E + (size_t)batch * 384 * DD;
  __shared__ __align__(16) _Float16 sB[64 * SB_STRIDE];
  const v8f zero = {};
  v8f acc[4] = {zero, zero, zero, zero};

  for (int kb = 0; kb < 384; kb += 32) {
    __syncthreads();
    stage_b_tile(Eb, DD, kb, col0, sB, threadIdx.x);
    __syncthreads();
    const v16h a = load_a_f16(A, 384, row0, kb, lane);
#pragma unroll
    for (int t = 0; t < 4; ++t)
      acc[t] = wmma16(a, load_b_staged(sB, t * 16, lane), acc[t]);
  }
  const int hs = lane >> 4, nloc = lane & 15;
#pragma unroll
  for (int t = 0; t < 4; ++t)
#pragma unroll
    for (int r = 0; r < 8; ++r)
      O[(size_t)(row0 + r + hs * 8) * DD + col0 + t * 16 + nloc] =
          (_Float16)acc[t][r];
}

// ---------------------------------------------------------------------------
// K5: fused compare FF + masked time-sum:
//   cmp = relu(concat([E f32, Att f16]) @ W_compare + b); out[b,:] += cmp*mask
// grid (8, 96, 2) x 256; in-register row reduction + global_atomic_add_f32.
// ---------------------------------------------------------------------------
__global__ void __launch_bounds__(256) compare_gemm_kernel(
    const float* __restrict__ Ep, const float* __restrict__ Eh,
    const _Float16* __restrict__ AttH, const _Float16* __restrict__ AttP,
    const float* __restrict__ Wc, const float* __restrict__ bc,
    const int* __restrict__ pm, const int* __restrict__ hm,
    float* __restrict__ cmpP, float* __restrict__ cmpH) {
  const float* E;
  const _Float16* Att;
  const int* mask;
  float* outSum;
  if (blockIdx.z == 0) { E = Ep; Att = AttH; mask = pm; outSum = cmpP; }
  else                 { E = Eh; Att = AttP; mask = hm; outSum = cmpH; }
  const int lane = threadIdx.x & 31;
  const int wave = threadIdx.x >> 5;
  const int row0 = blockIdx.y * 128 + wave * 16;
  const int col0 = blockIdx.x * 64;
  const int batch = row0 / 384;
  __shared__ __align__(16) _Float16 sB[64 * SB_STRIDE];
  const v8f zero = {};
  v8f acc[4] = {zero, zero, zero, zero};

  for (int kb = 0; kb < 2 * DD; kb += 32) {
    __syncthreads();
    stage_b_tile(Wc, DD, kb, col0, sB, threadIdx.x);
    __syncthreads();
    const v16h a = (kb < DD) ? load_a_f32(E, DD, row0, kb, lane)
                             : load_a_f16(Att, DD, row0, kb - DD, lane);
#pragma unroll
    for (int t = 0; t < 4; ++t)
      acc[t] = wmma16(a, load_b_staged(sB, t * 16, lane), acc[t]);
  }
  const int hs = lane >> 4, nloc = lane & 15;
#pragma unroll
  for (int t = 0; t < 4; ++t) {
    const int col = col0 + t * 16 + nloc;
    const float bv = bc[col];
    float partial = 0.0f;
#pragma unroll
    for (int r = 0; r < 8; ++r) {
      const int row = row0 + r + hs * 8;
      const float v = fmaxf(acc[t][r] + bv, 0.0f);
      partial += v * (float)mask[row];
    }
    partial += __shfl_xor(partial, 16, 32);  // merge rows 0-7 with 8-15
    if (hs == 0) atomicAdd(&outSum[batch * DD + col], partial);
  }
}

// ---------------------------------------------------------------------------
// K6: aggregate MLP + softmax over 3 labels. One 512-thread block per batch.
// ---------------------------------------------------------------------------
__global__ void __launch_bounds__(512) aggregate_kernel(
    const float* __restrict__ cmpP, const float* __restrict__ cmpH,
    const float* __restrict__ W1, const float* __restrict__ b1,
    const float* __restrict__ W2, const float* __restrict__ b2,
    float* __restrict__ out_logits, float* __restrict__ out_probs) {
  const int b = blockIdx.x;
  const int t = threadIdx.x;  // 0..511
  __shared__ float hid[512];
  __shared__ float lg[3];
  const float* ap = cmpP + b * 512;
  const float* ah = cmpH + b * 512;
  float acc = b1[t];
  for (int k = 0; k < 512; ++k) acc += ap[k] * W1[(size_t)k * 512 + t];
  for (int k = 0; k < 512; ++k) acc += ah[k] * W1[(size_t)(512 + k) * 512 + t];
  hid[t] = fmaxf(acc, 0.0f);
  __syncthreads();
  if (t < 3) {
    float a2 = b2[t];
    for (int k = 0; k < 512; ++k) a2 += hid[k] * W2[k * 3 + t];
    lg[t] = a2;
  }
  __syncthreads();
  if (t == 0) {
    const float m = fmaxf(lg[0], fmaxf(lg[1], lg[2]));
    const float e0 = __expf(lg[0] - m), e1 = __expf(lg[1] - m), e2 = __expf(lg[2] - m);
    const float inv = 1.0f / (e0 + e1 + e2);
    out_logits[b * 3 + 0] = lg[0];
    out_logits[b * 3 + 1] = lg[1];
    out_logits[b * 3 + 2] = lg[2];
    out_probs[b * 3 + 0] = e0 * inv;
    out_probs[b * 3 + 1] = e1 * inv;
    out_probs[b * 3 + 2] = e2 * inv;
  }
}

__global__ void zero_kernel(float* __restrict__ p, int n) {
  const int i = blockIdx.x * blockDim.x + threadIdx.x;
  if (i < n) p[i] = 0.0f;
}

// ---------------------------------------------------------------------------
extern "C" void kernel_launch(void* const* d_in, const int* in_sizes, int n_in,
                              void* d_out, int out_size, void* d_ws, size_t ws_size,
                              hipStream_t stream) {
  (void)in_sizes; (void)n_in; (void)out_size; (void)ws_size;
  const float* eP        = (const float*)d_in[0];   // [32,384,512]
  const float* eH        = (const float*)d_in[1];   // [32,384,512]
  const int*   pm        = (const int*)d_in[2];     // [32,384]
  const int*   hm        = (const int*)d_in[3];     // [32,384]
  const float* W_attend  = (const float*)d_in[4];   // [512,512]
  const float* b_attend  = (const float*)d_in[5];
  const float* W_compare = (const float*)d_in[6];   // [1024,512]
  const float* b_compare = (const float*)d_in[7];
  const float* W_agg1    = (const float*)d_in[8];   // [1024,512]
  const float* b_agg1    = (const float*)d_in[9];
  const float* W_agg2    = (const float*)d_in[10];  // [512,3]
  const float* b_agg2    = (const float*)d_in[11];

  // Workspace carve-up (256B aligned), ~84 MiB total.
  char* ws = (char*)d_ws;
  size_t off = 0;
  auto carve = [&](size_t bytes) {
    void* p = ws + off;
    off += (bytes + 255) & ~(size_t)255;
    return p;
  };
  const size_t M = (size_t)BB * PP;  // 12288 rows
  _Float16* projP = (_Float16*)carve(M * DD * 2);
  _Float16* projH = (_Float16*)carve(M * DD * 2);
  _Float16* p2h16 = (_Float16*)carve(M * HH * 2);
  _Float16* h2p16 = (_Float16*)carve(M * PP * 2);
  _Float16* attH  = (_Float16*)carve(M * DD * 2);
  _Float16* attP  = (_Float16*)carve(M * DD * 2);
  float*    sim   = (float*)carve(M * HH * 4);
  float*    cmpP  = (float*)carve((size_t)BB * DD * 4);
  float*    cmpH  = (float*)carve((size_t)BB * DD * 4);

  // Output layout: logits[32,3], probs[32,3], h2p[32,384,384], p2h[32,384,384]
  float* out        = (float*)d_out;
  float* out_logits = out;
  float* out_probs  = out + 96;
  float* out_h2p    = out + 192;
  float* out_p2h    = out + 192 + (size_t)BB * HH * PP;

  // cmpP/cmpH are contiguous in ws (each 64KB, 256B-aligned back-to-back).
  zero_kernel<<<(2 * BB * DD + 255) / 256, 256, 0, stream>>>(cmpP, 2 * BB * DD);

  attend_gemm_kernel<<<dim3(8, 96, 2), 256, 0, stream>>>(
      eP, eH, W_attend, b_attend, projP, projH);

  sim_gemm_kernel<<<dim3(6, 3, 32), 256, 0, stream>>>(projP, projH, sim);

  masked_softmax_kernel<<<BB * PP, 128, 0, stream>>>(sim, hm, out_p2h, p2h16, 0);
  masked_softmax_kernel<<<BB * HH, 128, 0, stream>>>(sim, pm, out_h2p, h2p16, 1);

  attn_gemm_kernel<<<dim3(8, 96, 2), 256, 0, stream>>>(
      p2h16, h2p16, eP, eH, attH, attP);

  compare_gemm_kernel<<<dim3(8, 96, 2), 256, 0, stream>>>(
      eP, eH, attH, attP, W_compare, b_compare, pm, hm, cmpP, cmpH);

  aggregate_kernel<<<BB, 512, 0, stream>>>(
      cmpP, cmpH, W_agg1, b_agg1, W_agg2, b_agg2, out_logits, out_probs);
}
